// IoUPredIG_3659312136532
// MI455X (gfx1250) — compile-verified, matrix operand order checked by reading, not run on hardware
//
#include <hip/hip_runtime.h>
#include <stdint.h>

#define NUM_CLASSES 80
#define NUM_ANCHORS 9
#define CIN 256
#define WAVES 8
#define CAP 1024
#define SORT_N 4096
#define KEEP 1024
#define MAX_DET 100
#define TOT_HWA 179550

typedef _Float16 v16h __attribute__((ext_vector_type(16)));
typedef _Float16 v8h  __attribute__((ext_vector_type(8)));
typedef float    v8f  __attribute__((ext_vector_type(8)));

static __device__ __forceinline__ float sigmoidf_(float x) {
    return 1.0f / (1.0f + __expf(-x));
}

// ---------------------------------------------------------------------------
// Weight conversion: subnet (4,256,256,3,3) f32 -> f16 [layer][tap][cout][cin]
// ---------------------------------------------------------------------------
__global__ void cvt_subnet_w_kernel(const float* __restrict__ w, _Float16* __restrict__ o) {
    const int total = 4 * 256 * 256 * 9;
    for (int idx = blockIdx.x * blockDim.x + threadIdx.x; idx < total;
         idx += gridDim.x * blockDim.x) {
        int kx = idx % 3; int t = idx / 3;
        int ky = t % 3;   t /= 3;
        int ci = t % 256; t /= 256;
        int co = t % 256; int layer = t / 256;
        o[((size_t)(layer * 9 + ky * 3 + kx) * 256 + co) * 256 + ci] = (_Float16)w[idx];
    }
}

// head (720,256,3,3) f32 -> f16 [tap][cout][cin]
__global__ void cvt_head_w_kernel(const float* __restrict__ w, _Float16* __restrict__ o) {
    const int total = 720 * 256 * 9;
    for (int idx = blockIdx.x * blockDim.x + threadIdx.x; idx < total;
         idx += gridDim.x * blockDim.x) {
        int kx = idx % 3; int t = idx / 3;
        int ky = t % 3;   t /= 3;
        int ci = t % 256; int co = t / 256;
        o[((size_t)(ky * 3 + kx) * 720 + co) * 256 + ci] = (_Float16)w[idx];
    }
}

// NCHW f32 (one image) -> NHWC f16
__global__ void cvt_feat_kernel(const float* __restrict__ in, _Float16* __restrict__ o, int HW) {
    const int total = 256 * HW;
    for (int idx = blockIdx.x * blockDim.x + threadIdx.x; idx < total;
         idx += gridDim.x * blockDim.x) {
        int hw = idx % HW;
        int c  = idx / HW;
        o[(size_t)hw * 256 + c] = (_Float16)in[idx];
    }
}

// ---------------------------------------------------------------------------
// Implicit-GEMM 3x3 SAME conv, Cin=256, f16 in (NHWC), WMMA f32 accumulate.
// Workgroup: 8 waves, 16-pixel row tile. Each wave register-blocks TWO 16-cout
// tiles (16x32 output): 72 flattened (tap,ck) steps, explicitly double-
// buffered (prefetch it+1's A/B fragments before it's two WMMAs) so the
// FIFO load counters keep loads in flight across the WMMA pair.
// __launch_bounds__(256,1) releases the occupancy register cap so both
// B-fragment buffers stay live (avoids the s_wait_loadcnt 0x0 full drains).
// ---------------------------------------------------------------------------
__global__ __launch_bounds__(256, 1)
void conv3x3_wmma_kernel(const _Float16* __restrict__ in, const _Float16* __restrict__ wt,
                         const float* __restrict__ bias, _Float16* __restrict__ outH,
                         float* __restrict__ outHead,
                         int H, int W, int Cout, int tilesPerRow, int doRelu) {
    __shared__ _Float16 sA[3 * 18 * CIN];  // 27.6 KB halo tile
    const int h     = blockIdx.x / tilesPerRow;
    const int wBase = (blockIdx.x % tilesPerRow) * 16;

    // Cooperative stage: rows h-1..h+1, pixels wBase-1..wBase+16, all 256 ch.
    for (int t = threadIdx.x; t < 3 * 18 * (CIN / 8); t += 256) {
        int slot  = t >> 5;           // pixel slot (0..53)
        int cpart = t & 31;           // 8-channel chunk
        int r = slot / 18, p = slot - r * 18;
        int gh = h - 1 + r, gw = wBase - 1 + p;
        v8h val = {};
        if (gh >= 0 && gh < H && gw >= 0 && gw < W)
            val = *(const v8h*)(in + (size_t)(gh * W + gw) * CIN + cpart * 8);
        *(v8h*)(sA + slot * CIN + cpart * 8) = val;
    }
    __syncthreads();

    const int lane = threadIdx.x & 31;
    const int wave = threadIdx.x >> 5;
    const int hi   = lane >> 4;   // K-half selector per ISA 16-bit fragment layout
    const int l15  = lane & 15;   // A row (pixel) == B column (cout)
    const int cout0 = (blockIdx.y * WAVES + wave) * 32;
    const int cout1 = cout0 + 16;
    if (cout0 >= Cout) return;
    const bool tile1 = (cout1 < Cout);
    // Clamp second tile's weight rows in-bounds; its results are discarded.
    const int wrow1c = tile1 ? (cout1 + l15) : (Cout - 1);

    // Base pointers; per-step addressing folds to constant immediate offsets.
    const _Float16* aBase  = sA + l15 * CIN + hi * 8;
    const _Float16* wBase0 = wt + (size_t)(cout0 + l15) * CIN + hi * 8;
    const _Float16* wBase1 = wt + (size_t)wrow1c * CIN + hi * 8;
    const size_t    wTapStride = (size_t)Cout * CIN;

    union Frag { v16h v; v8h p[2]; };
    Frag afb[2], b0b[2], b1b[2];

    auto LD = [&](int it, int buf) {
        const int tap = it >> 3, ck = it & 7;
        const int ky = tap / 3, kx = tap - ky * 3;
        const _Float16* ar = aBase + (ky * 18 + kx) * CIN + ck * 32;
        const _Float16* w0 = wBase0 + tap * wTapStride + ck * 32;
        const _Float16* w1 = wBase1 + tap * wTapStride + ck * 32;
        afb[buf].p[0] = *(const v8h*)ar;          // K = hi*8 .. +7
        afb[buf].p[1] = *(const v8h*)(ar + 16);   // K = 16+hi*8 .. +7
        b0b[buf].p[0] = *(const v8h*)w0;
        b0b[buf].p[1] = *(const v8h*)(w0 + 16);
        b1b[buf].p[0] = *(const v8h*)w1;
        b1b[buf].p[1] = *(const v8h*)(w1 + 16);
    };

    v8f acc0 = {}, acc1 = {};
    LD(0, 0);
#pragma unroll
    for (int it = 0; it < 72; ++it) {
        const int cur = it & 1, nxt = cur ^ 1;
        if (it + 1 < 72) LD(it + 1, nxt);          // prefetch next fragments
        acc0 = __builtin_amdgcn_wmma_f32_16x16x32_f16(
            false, afb[cur].v, false, b0b[cur].v, (short)0, acc0, false, false);
        acc1 = __builtin_amdgcn_wmma_f32_16x16x32_f16(
            false, afb[cur].v, false, b1b[cur].v, (short)0, acc1, false, false);
    }

    const float bv0 = bias[cout0 + l15];
    const float bv1 = bias[tile1 ? (cout1 + l15) : (Cout - 1)];
    if (outHead == nullptr) {
#pragma unroll
        for (int r = 0; r < 8; ++r) {
            int m  = r + hi * 8;   // C/D layout: VGPR r holds M=r (lanes 0-15) / M=r+8
            int gw = wBase + m;
            if (gw < W) {
                float v0 = acc0[r] + bv0;
                if (doRelu) v0 = fmaxf(v0, 0.0f);
                outH[(size_t)(h * W + gw) * CIN + cout0 + l15] = (_Float16)v0;
                if (tile1) {
                    float v1 = acc1[r] + bv1;
                    if (doRelu) v1 = fmaxf(v1, 0.0f);
                    outH[(size_t)(h * W + gw) * CIN + cout1 + l15] = (_Float16)v1;
                }
            }
        }
    } else {
        const int a0 = cout0 / NUM_CLASSES, k0 = cout0 % NUM_CLASSES;  // 80=5*16: tile
        const int a1 = cout1 / NUM_CLASSES, k1 = cout1 % NUM_CLASSES;  // stays in anchor
#pragma unroll
        for (int r = 0; r < 8; ++r) {
            int m  = r + hi * 8;
            int gw = wBase + m;
            if (gw < W) {
                size_t px = (size_t)(h * W + gw) * NUM_ANCHORS;
                outHead[(px + a0) * NUM_CLASSES + k0 + l15] = acc0[r] + bv0;
                if (tile1)
                    outHead[(px + a1) * NUM_CLASSES + k1 + l15] = acc1[r] + bv1;
            }
        }
    }
}

// ---------------------------------------------------------------------------
// Candidate scoring: sigmoid(cls)*sigmoid(iou) > 0.05 -> append (score,key)
// ---------------------------------------------------------------------------
__global__ void zero_cnt_kernel(int* cnt) {
    if (threadIdx.x < 8) cnt[threadIdx.x] = 0;
}

__global__ void score_kernel(const float* __restrict__ cls, const float* __restrict__ iou,
                             int n, int level, int* __restrict__ cnt,
                             uint2* __restrict__ cand) {
    int i = blockIdx.x * blockDim.x + threadIdx.x;
    if (i >= n) return;
    float s = sigmoidf_(cls[i]) * sigmoidf_(iou[i]);
    if (s > 0.05f) {
        int pos = atomicAdd(cnt, 1);
        if (pos < CAP) {
            unsigned anchor = (unsigned)(i / NUM_CLASSES);
            unsigned c      = (unsigned)(i % NUM_CLASSES);
            uint2 e;
            e.x = __float_as_uint(s);
            e.y = ((unsigned)level << 25) | (anchor << 7) | c;
            cand[pos] = e;
        }
    }
}

// ---------------------------------------------------------------------------
// Per-image: bitonic sort 4096 candidates by score, class-aware greedy NMS
// over top 1024, emit top-100 (boxes, scores, classes).
// ---------------------------------------------------------------------------
__global__ __launch_bounds__(256)
void nms_kernel(const int* __restrict__ cnt, const uint2* __restrict__ cand,
                const float* __restrict__ box0, const float* __restrict__ box1,
                const float* __restrict__ box2, float* __restrict__ out, int b) {
    __shared__ float    ss[SORT_N];
    __shared__ unsigned kk[SORT_N];
    __shared__ float    bx[KEEP * 4];
    __shared__ int      cc[KEEP];
    __shared__ int      sup[KEEP];
    const int tid = threadIdx.x;
    const int hwa0 = 136800, hwa1 = 34200, hwa2 = 8550;

    for (int i = tid; i < SORT_N; i += 256) {
        float s = -1.0f; unsigned k = 0;
        if (i < 3 * CAP) {
            int l = i / CAP, j = i % CAP;
            int n = cnt[b * 3 + l]; if (n > CAP) n = CAP;
            if (j < n) {
                uint2 e = cand[((size_t)b * 3 + l) * CAP + j];
                s = __uint_as_float(e.x); k = e.y;
            }
        }
        ss[i] = s; kk[i] = k;
    }
    __syncthreads();

    // bitonic sort, descending by score
    for (unsigned kstep = 2; kstep <= SORT_N; kstep <<= 1) {
        for (unsigned j = kstep >> 1; j > 0; j >>= 1) {
            for (unsigned i = tid; i < SORT_N; i += 256) {
                unsigned ixj = i ^ j;
                if (ixj > i) {
                    bool descBlock = ((i & kstep) == 0);
                    float a = ss[i], c2 = ss[ixj];
                    bool doSwap = descBlock ? (a < c2) : (a > c2);
                    if (doSwap) {
                        ss[i] = c2; ss[ixj] = a;
                        unsigned t = kk[i]; kk[i] = kk[ixj]; kk[ixj] = t;
                    }
                }
            }
            __syncthreads();
        }
    }

    // gather boxes / classes for top KEEP
    for (int i = tid; i < KEEP; i += 256) {
        sup[i] = 0; cc[i] = -1;
        if (ss[i] > 0.0f) {
            unsigned k = kk[i];
            int c = k & 127, anchor = (k >> 7) & 0x3FFFF, l = (int)(k >> 25);
            const float* bp  = (l == 0) ? box0 : (l == 1 ? box1 : box2);
            int          hwa = (l == 0) ? hwa0 : (l == 1 ? hwa1 : hwa2);
            const float* p   = bp + ((size_t)b * hwa + anchor) * 4;
            bx[i * 4 + 0] = p[0]; bx[i * 4 + 1] = p[1];
            bx[i * 4 + 2] = p[2]; bx[i * 4 + 3] = p[3];
            cc[i] = c;
        }
    }
    __syncthreads();

    // greedy class-aware NMS (class-offset trick == same-class-only IoU)
    for (int i = 0; i < KEEP; ++i) {
        bool alive = (ss[i] > 0.0f) && (sup[i] == 0);
        if (alive) {
            float x0 = bx[i * 4], y0 = bx[i * 4 + 1], x1 = bx[i * 4 + 2], y1 = bx[i * 4 + 3];
            float areaI = (x1 - x0) * (y1 - y0);
            int ci = cc[i];
            for (int j = i + 1 + tid; j < KEEP; j += 256) {
                if (sup[j] == 0 && cc[j] == ci && ss[j] > 0.0f) {
                    float u0 = fmaxf(x0, bx[j * 4]),     v0 = fmaxf(y0, bx[j * 4 + 1]);
                    float u1 = fminf(x1, bx[j * 4 + 2]), v1 = fminf(y1, bx[j * 4 + 3]);
                    float iw = fmaxf(u1 - u0, 0.0f), ih = fmaxf(v1 - v0, 0.0f);
                    float inter = iw * ih;
                    float areaJ = (bx[j * 4 + 2] - bx[j * 4]) * (bx[j * 4 + 3] - bx[j * 4 + 1]);
                    if (inter / (areaI + areaJ - inter + 1e-9f) > 0.5f) sup[j] = 1;
                }
            }
        }
        __syncthreads();
    }

    if (tid == 0) {
        const size_t base = (size_t)2 * TOT_HWA * NUM_CLASSES;
        float* dbox = out + base + (size_t)b * MAX_DET * 4;
        float* dsc  = out + base + 2 * MAX_DET * 4 + (size_t)b * MAX_DET;
        float* dcl  = out + base + 2 * MAX_DET * 4 + 2 * MAX_DET + (size_t)b * MAX_DET;
        int n = 0;
        for (int i = 0; i < KEEP && n < MAX_DET; ++i) {
            if (ss[i] > 0.0f && sup[i] == 0) {
                dbox[n * 4 + 0] = bx[i * 4 + 0]; dbox[n * 4 + 1] = bx[i * 4 + 1];
                dbox[n * 4 + 2] = bx[i * 4 + 2]; dbox[n * 4 + 3] = bx[i * 4 + 3];
                dsc[n] = ss[i]; dcl[n] = (float)cc[i]; ++n;
            }
        }
        for (; n < MAX_DET; ++n) {
            dbox[n * 4 + 0] = 0.0f; dbox[n * 4 + 1] = 0.0f;
            dbox[n * 4 + 2] = 0.0f; dbox[n * 4 + 3] = 0.0f;
            dsc[n] = 0.0f; dcl[n] = 0.0f;
        }
    }
}

// ---------------------------------------------------------------------------
extern "C" void kernel_launch(void* const* d_in, const int* in_sizes, int n_in,
                              void* d_out, int out_size, void* d_ws, size_t ws_size,
                              hipStream_t stream) {
    (void)in_sizes; (void)n_in; (void)out_size; (void)ws_size;
    const int Hs[3] = {100, 50, 25}, Wsz[3] = {152, 76, 38};
    const int HWA[3]  = {136800, 34200, 8550};
    const int AOFF[3] = {0, 136800, 171000};

    const float* feat[3] = {(const float*)d_in[0], (const float*)d_in[3], (const float*)d_in[6]};
    const float* cls[3]  = {(const float*)d_in[1], (const float*)d_in[4], (const float*)d_in[7]};
    const float* box[3]  = {(const float*)d_in[2], (const float*)d_in[5], (const float*)d_in[8]};
    const float* subW  = (const float*)d_in[9];
    const float* subB  = (const float*)d_in[10];
    const float* headW = (const float*)d_in[11];
    const float* headB = (const float*)d_in[12];
    float* out = (float*)d_out;

    char* ws = (char*)d_ws;
    size_t off = 0;
    auto alloc = [&](size_t bytes) {
        void* p = ws + off;
        off = (off + bytes + 255) & ~(size_t)255;
        return p;
    };
    _Float16* wSub  = (_Float16*)alloc((size_t)4 * 9 * 256 * 256 * 2);
    _Float16* wHead = (_Float16*)alloc((size_t)9 * 720 * 256 * 2);
    _Float16* actIn = (_Float16*)alloc((size_t)15200 * 256 * 2);
    _Float16* actA  = (_Float16*)alloc((size_t)15200 * 256 * 2);
    _Float16* actB  = (_Float16*)alloc((size_t)15200 * 256 * 2);
    int*      cnt   = (int*)alloc(256);
    uint2*    cand  = (uint2*)alloc((size_t)2 * 3 * CAP * sizeof(uint2));

    cvt_subnet_w_kernel<<<4096, 256, 0, stream>>>(subW, wSub);
    cvt_head_w_kernel<<<4096, 256, 0, stream>>>(headW, wHead);
    zero_cnt_kernel<<<1, 32, 0, stream>>>(cnt);

    for (int b = 0; b < 2; ++b) {
        for (int l = 0; l < 3; ++l) {
            const int H = Hs[l], W = Wsz[l], HW = H * W;
            const int tpr = (W + 15) / 16;
            cvt_feat_kernel<<<(256 * HW + 255) / 256, 256, 0, stream>>>(
                feat[l] + (size_t)b * 256 * HW, actIn, HW);

            dim3 gs(H * tpr, 1);   // 256 couts = 8 waves x 32
            conv3x3_wmma_kernel<<<gs, 256, 0, stream>>>(
                actIn, wSub + (size_t)0 * 9 * 256 * 256, subB + 0 * 256, actA, nullptr, H, W, 256, tpr, 1);
            conv3x3_wmma_kernel<<<gs, 256, 0, stream>>>(
                actA, wSub + (size_t)1 * 9 * 256 * 256, subB + 1 * 256, actB, nullptr, H, W, 256, tpr, 1);
            conv3x3_wmma_kernel<<<gs, 256, 0, stream>>>(
                actB, wSub + (size_t)2 * 9 * 256 * 256, subB + 2 * 256, actA, nullptr, H, W, 256, tpr, 1);
            conv3x3_wmma_kernel<<<gs, 256, 0, stream>>>(
                actA, wSub + (size_t)3 * 9 * 256 * 256, subB + 3 * 256, actB, nullptr, H, W, 256, tpr, 1);

            float* iouDst = out + (size_t)b * TOT_HWA * NUM_CLASSES + (size_t)AOFF[l] * NUM_CLASSES;
            dim3 gh(H * tpr, 3);   // 720 couts -> 23 x 32-wide tiles, 8 waves/wg
            conv3x3_wmma_kernel<<<gh, 256, 0, stream>>>(
                actB, wHead, headB, nullptr, iouDst, H, W, 720, tpr, 0);

            const int nelem = HWA[l] * NUM_CLASSES;
            score_kernel<<<(nelem + 255) / 256, 256, 0, stream>>>(
                cls[l] + (size_t)b * nelem, iouDst, nelem, l,
                cnt + b * 3 + l, cand + ((size_t)b * 3 + l) * CAP);
        }
    }
    for (int b = 0; b < 2; ++b)
        nms_kernel<<<1, 256, 0, stream>>>(cnt, cand, box[0], box[1], box[2], out, b);
}
